// InterLayer_41575283426070
// MI455X (gfx1250) — compile-verified, matrix operand order checked by reading, not run on hardware
//
#include <hip/hip_runtime.h>
#include <hip/hip_bf16.h>

// ---------------- problem constants (from reference setup_inputs) ------------
#define SIZE_B  2048      // batch / #text sequences
#define TEXT_S  96        // text sequence length
#define DMODEL  256       // model dim
#define NHEADS  8
#define HDIM    32        // DMODEL / NHEADS
#define NNODES  32768     // graph nodes
#define KNBR    32        // neighbors per node
#define E2      65536     // SIZE_B * KNBR (directed edges, one direction)
#define EFULL   131072    // 2 * E2 (full edge_index columns)
#define GS      33        // graph attention sequence length (1 + KNBR)
#define GST     48        // GS padded to multiple of 16

typedef __bf16 bf16_t;
typedef __attribute__((ext_vector_type(16))) __bf16 v16bf;
typedef __attribute__((ext_vector_type(4)))  __bf16 v4bf;
typedef __attribute__((ext_vector_type(8)))  float  v8f;

// ---------------------------------------------------------------------------
// WMMA helpers (CDNA5 wave32, 16x16x32 bf16 -> f32)
// A-frag (16x32, 16-bit): lane half = lane>>4, row = lane&15.
//   element j holds K = half*8 + (j&7) + 16*(j>>3)  (per ISA 16-bit A layout)
// B-frag uses the same per-lane K packing, lane = column.
// C/D frag: vgpr r -> row = r + 8*(lane>>4), col = lane&15.
// ---------------------------------------------------------------------------
__device__ __forceinline__ v8f wmma_bf16(v16bf a, v16bf b, v8f c) {
  return __builtin_amdgcn_wmma_f32_16x16x32_bf16(
      /*neg_a=*/false, a, /*neg_b=*/false, b,
      /*c_mod=*/(short)0, c, /*reuse_a=*/false, /*reuse_b=*/false);
}

// Rows of `base` are contiguous runs of K (row-major, stride ld in bf16 elems).
__device__ __forceinline__ v16bf frag_row16(const bf16_t* base, int row0,
                                            int ld, int lane) {
  const int half = lane >> 4, r = lane & 15;
  const bf16_t* p = base + (size_t)(row0 + r) * ld;
  v16bf f;
#pragma unroll
  for (int j = 0; j < 16; ++j) {
    int k = (half << 3) + (j & 7) + ((j >> 3) << 4);
    f[j] = p[k];
  }
  return f;
}

// B fragment from a K-major (row = K index) tile: lane = output column.
__device__ __forceinline__ v16bf frag_col16(const bf16_t* base, int krow0,
                                            int col0, int ld, int lane) {
  const int half = lane >> 4, c = lane & 15;
  v16bf f;
#pragma unroll
  for (int j = 0; j < 16; ++j) {
    int k = (half << 3) + (j & 7) + ((j >> 3) << 4);
    f[j] = base[(size_t)(krow0 + k) * ld + col0 + c];
  }
  return f;
}

__device__ __forceinline__ void cvt_store4(bf16_t* dst, float4 v) {
  v4bf b;
  b[0] = (bf16_t)v.x; b[1] = (bf16_t)v.y;
  b[2] = (bf16_t)v.z; b[3] = (bf16_t)v.w;
  *(v4bf*)dst = b;                       // 8-byte ds_store
}

// ---------------------------------------------------------------------------
// Generic GEMM: C[M,N] = A[M,K] (row stride lda) @ W[N,K]^T + bias[N]
// fp32 in / fp32 out, bf16 WMMA math. Block tile 128x64, 8 waves (4x2),
// each wave 32x32 (2x2 frags). Requires M%128==0, N%64==0, K%32==0
// (true for every call site). Double-buffered LDS, loads issued ahead of
// the barrier so they overlap the WMMAs of the current tile.
// ---------------------------------------------------------------------------
__global__ __launch_bounds__(256)
void gemm_bf16_wmma(const float* __restrict__ A, int lda,
                    const float* __restrict__ W,
                    const float* __restrict__ bias,
                    float* __restrict__ C, int ldc,
                    int M, int N, int K) {
  __shared__ bf16_t As[2][128][40];   // +8 pad vs 32 to dodge bank conflicts
  __shared__ bf16_t Bs[2][64][40];
  const int tid  = threadIdx.x;
  const int lane = tid & 31, wave = tid >> 5;
  const int wm = wave & 3, wn = wave >> 2;
  const int m0 = blockIdx.y * 128, n0 = blockIdx.x * 64;

  // per-thread staging map: float4 granules of a 32-wide K slab
  // A tile: 128 rows * 8 granules = 1024 -> 4 granules/thread
  // W tile:  64 rows * 8 granules =  512 -> 2 granules/thread
  int ar[4], ac[4], br[2], bc[2];
#pragma unroll
  for (int i = 0; i < 4; ++i) {
    int idx4 = tid + i * 256;
    ar[i] = idx4 >> 3; ac[i] = (idx4 & 7) << 2;
  }
#pragma unroll
  for (int i = 0; i < 2; ++i) {
    int idx4 = tid + i * 256;
    br[i] = idx4 >> 3; bc[i] = (idx4 & 7) << 2;
  }

  // prologue: stage tile kb=0 into buffer 0
#pragma unroll
  for (int i = 0; i < 4; ++i) {
    float4 v = *(const float4*)(A + (size_t)(m0 + ar[i]) * lda + ac[i]);
    cvt_store4(&As[0][ar[i]][ac[i]], v);
  }
#pragma unroll
  for (int i = 0; i < 2; ++i) {
    float4 v = *(const float4*)(W + (size_t)(n0 + br[i]) * K + bc[i]);
    cvt_store4(&Bs[0][br[i]][bc[i]], v);
  }

  v8f acc[2][2] = {};
  int buf = 0;

  for (int kb = 0; kb < K; kb += 32) {
    const bool nxt = (kb + 32) < K;
    float4 an[4], bn[2];
    if (nxt) {                       // issue next tile's loads (stay in flight)
#pragma unroll
      for (int i = 0; i < 4; ++i)
        an[i] = *(const float4*)(A + (size_t)(m0 + ar[i]) * lda + kb + 32 + ac[i]);
#pragma unroll
      for (int i = 0; i < 2; ++i)
        bn[i] = *(const float4*)(W + (size_t)(n0 + br[i]) * K + kb + 32 + bc[i]);
    }
    __syncthreads();                 // current buffer ready / prev reads done

    v16bf a0 = frag_row16(&As[buf][0][0], wm * 32,      40, lane);
    v16bf a1 = frag_row16(&As[buf][0][0], wm * 32 + 16, 40, lane);
    v16bf b0 = frag_row16(&Bs[buf][0][0], wn * 32,      40, lane);
    v16bf b1 = frag_row16(&Bs[buf][0][0], wn * 32 + 16, 40, lane);
    acc[0][0] = wmma_bf16(a0, b0, acc[0][0]);
    acc[0][1] = wmma_bf16(a0, b1, acc[0][1]);
    acc[1][0] = wmma_bf16(a1, b0, acc[1][0]);
    acc[1][1] = wmma_bf16(a1, b1, acc[1][1]);

    if (nxt) {                       // drain loads into the other buffer
#pragma unroll
      for (int i = 0; i < 4; ++i) cvt_store4(&As[buf ^ 1][ar[i]][ac[i]], an[i]);
#pragma unroll
      for (int i = 0; i < 2; ++i) cvt_store4(&Bs[buf ^ 1][br[i]][bc[i]], bn[i]);
    }
    buf ^= 1;
  }

  const int half = lane >> 4, nn = lane & 15;
#pragma unroll
  for (int i = 0; i < 2; ++i)
#pragma unroll
    for (int j = 0; j < 2; ++j) {
      int n = n0 + wn * 32 + j * 16 + nn;
      float bv = (bias != nullptr) ? bias[n] : 0.0f;
#pragma unroll
      for (int r = 0; r < 8; ++r) {
        int m = m0 + wm * 32 + i * 16 + r + half * 8;
        C[(size_t)m * ldc + n] = acc[i][j][r] + bv;
      }
    }
}

// ---------------------------------------------------------------------------
// Fused MHA: one block per batch element, loops heads. qkv is (B*S, 3*256).
// Scores and probs never leave LDS. ctx written as (B*S, 256).
// WRITE_ATTN: accumulate head-mean attn into attn_out (B,S,S).
// HAS_MASK: padding_mask (B, S-1) u8; key 0 never masked.
// ---------------------------------------------------------------------------
template <int S, int ST, bool WRITE_ATTN, bool HAS_MASK>
__global__ __launch_bounds__(256)
void mha_kernel(const float* __restrict__ qkv,
                const unsigned char* __restrict__ mask,
                float* __restrict__ ctx,
                float* __restrict__ attn_out) {
  constexpr int SCP = ST + 4;               // fp32 score row stride
  constexpr int NT  = ST / 16;
  __shared__ bf16_t Qs[ST][40];
  __shared__ bf16_t Ks[ST][40];
  __shared__ bf16_t Vs[ST][40];
  __shared__ float  Sc[ST][SCP];            // scores; reused as bf16 probs

  const int b = blockIdx.x;
  const int tid = threadIdx.x, lane = tid & 31, wave = tid >> 5;
  const float scale = 0.17677669529663687f; // 1/sqrt(32)

  for (int h = 0; h < NHEADS; ++h) {
    __syncthreads();
    // ---- stage Q/K/V head slices into LDS (float4 loads, bf16x4 stores) ---
    for (int idx4 = tid; idx4 < ST * 8; idx4 += 256) {
      int s = idx4 >> 3, d4 = (idx4 & 7) << 2;
      float4 q = {0.f, 0.f, 0.f, 0.f}, k = q, v = q;
      if (s < S) {
        const float* row = qkv + ((size_t)b * S + s) * (3 * DMODEL);
        q = *(const float4*)(row + h * HDIM + d4);
        k = *(const float4*)(row + DMODEL + h * HDIM + d4);
        v = *(const float4*)(row + 2 * DMODEL + h * HDIM + d4);
      }
      cvt_store4(&Qs[s][d4], q);
      cvt_store4(&Ks[s][d4], k);
      cvt_store4(&Vs[s][d4], v);
    }
    __syncthreads();

    // ---- scores = Q @ K^T * scale, with key mask / padding kill ----------
    for (int t = wave; t < NT * NT; t += 8) {
      int qt = t / NT, kt = t % NT;
      v16bf aq = frag_row16(&Qs[0][0], qt * 16, 40, lane);
      v16bf bk = frag_row16(&Ks[0][0], kt * 16, 40, lane);
      v8f acc = {};
      acc = wmma_bf16(aq, bk, acc);
      int half = lane >> 4, n = lane & 15;
      int kk = kt * 16 + n;
      bool kill = (kk >= S);
      if (HAS_MASK && !kill && kk > 0)
        kill = (mask[(size_t)b * (S - 1) + kk - 1] != 0);
#pragma unroll
      for (int r = 0; r < 8; ++r) {
        int q = qt * 16 + r + half * 8;
        Sc[q][kk] = kill ? -1.0e9f : acc[r] * scale;
      }
    }
    __syncthreads();

    // ---- softmax (one row per thread) ------------------------------------
    if (tid < S) {
      float* row = &Sc[tid][0];
      float mx = -3.0e38f;
      for (int k = 0; k < ST; ++k) mx = fmaxf(mx, row[k]);
      float sum = 0.0f;
      for (int k = 0; k < ST; ++k) { float e = __expf(row[k] - mx); row[k] = e; sum += e; }
      float inv = 1.0f / sum;
      for (int k = 0; k < ST; ++k) row[k] *= inv;
    }
    __syncthreads();

    // ---- head-mean attention map straight to global (L2-resident) --------
    if (WRITE_ATTN) {
      float* am = attn_out + (size_t)b * S * S;
      for (int idx = tid; idx < S * S; idx += 256) {
        int q = idx / S, k = idx % S;
        float v = Sc[q][k] * 0.125f;     // mean over 8 heads
        if (h == 0) am[idx] = v; else am[idx] += v;
      }
      __syncthreads();
    }

    // ---- convert prob rows fp32 -> bf16 in place (ascending is safe) -----
    if (tid < ST) {
      const float* rowf = &Sc[tid][0];
      bf16_t* rowb = (bf16_t*)&Sc[tid][0];
      if (tid < S) {
        for (int k = 0; k < ST; ++k) { float v = rowf[k]; rowb[k] = (bf16_t)v; }
      } else {
        for (int k = 0; k < ST; ++k) rowb[k] = (bf16_t)0.0f;
      }
    }
    __syncthreads();

    // ---- ctx = P @ V  (M=ST, K=ST, N=32) ---------------------------------
    const bf16_t* Pb = (const bf16_t*)&Sc[0][0];
    const int pld = 2 * SCP;                 // bf16 row stride of prob matrix
    for (int t = wave; t < NT * 2; t += 8) {
      int qt = t >> 1, dt = t & 1;
      v8f acc = {};
      for (int kb = 0; kb < ST; kb += 32) {
        v16bf ap = frag_row16(Pb + kb, qt * 16, pld, lane);
        v16bf bv = frag_col16(&Vs[0][0], kb, dt * 16, 40, lane);
        acc = wmma_bf16(ap, bv, acc);
      }
      int half = lane >> 4, d = dt * 16 + (lane & 15);
#pragma unroll
      for (int r = 0; r < 8; ++r) {
        int q = qt * 16 + r + half * 8;
        if (q < S)
          ctx[((size_t)b * S + q) * DMODEL + h * HDIM + d] = acc[r];
      }
    }
  }
}

// ---------------------------------------------------------------------------
// GCN helper kernels
// ---------------------------------------------------------------------------
__global__ void deg_init_kernel(float* deg) {
  int i = blockIdx.x * 256 + threadIdx.x;
  if (i < NNODES) deg[i] = 1.0f;            // the "+ 1.0" in the reference
}

__global__ void deg_count_kernel(const int* __restrict__ ei, float* deg) {
  int e = blockIdx.x * 256 + threadIdx.x;
  if (e < EFULL) atomicAdd(&deg[ei[EFULL + e]], 1.0f);   // dst row
}

__global__ void res_base_kernel(const float* __restrict__ xw,
                                const float* __restrict__ deg,
                                const float* __restrict__ bias,
                                float* __restrict__ res) {
  size_t idx = (size_t)blockIdx.x * 256 + threadIdx.x;
  if (idx < (size_t)NNODES * DMODEL) {
    int i = (int)(idx >> 8), c = (int)(idx & 255);
    res[idx] = xw[idx] / deg[i] + bias[c];   // dn*dn == 1/deg
  }
}

__global__ __launch_bounds__(256)
void scatter_kernel(const int* __restrict__ ei, const float* __restrict__ xw,
                    const float* __restrict__ deg, float* __restrict__ res) {
  int gw   = (blockIdx.x * 256 + threadIdx.x) >> 5;  // one wave per edge
  int lane = threadIdx.x & 31;
  if (gw >= EFULL) return;
  int s = ei[gw], d = ei[EFULL + gw];
  float coef = rsqrtf(deg[s]) * rsqrtf(deg[d]);
  const float* src = xw + (size_t)s * DMODEL;
  float* dst = res + (size_t)d * DMODEL;
#pragma unroll
  for (int i = 0; i < 8; ++i) {
    int c = lane + i * 32;
    atomicAdd(&dst[c], src[c] * coef);
  }
}

__global__ void gather_kernel(const float* __restrict__ res,
                              const int* __restrict__ ei,
                              float* __restrict__ neigh) {
  size_t idx = (size_t)blockIdx.x * 256 + threadIdx.x;
  if (idx >= (size_t)SIZE_B * GS * DMODEL) return;
  int c = (int)(idx & 255);
  int row = (int)(idx >> 8);
  int i = row / GS, j = row % GS;
  // src[:E2] is already sorted => stable argsort is identity => nbr = dst[:E2]
  int s = (j == 0) ? i : ei[EFULL + (i * KNBR + (j - 1))];
  neigh[idx] = res[(size_t)s * DMODEL + c];
}

__global__ void copy_tail_kernel(const float* __restrict__ res,
                                 float* __restrict__ gout) {
  size_t idx = (size_t)blockIdx.x * 256 + threadIdx.x;
  size_t n = (size_t)(NNODES - SIZE_B) * DMODEL;
  if (idx < n) gout[(size_t)SIZE_B * DMODEL + idx] =
               res[(size_t)SIZE_B * DMODEL + idx];
}

// ---------------------------------------------------------------------------
// Interlayer gating: one wave (32 lanes) per batch row of 256 features.
// ---------------------------------------------------------------------------
__global__ __launch_bounds__(256)
void gate_kernel(float* __restrict__ tout, const float* __restrict__ gh,
                 const float* __restrict__ ts, const float* __restrict__ gs,
                 float* __restrict__ gout) {
  int b = (blockIdx.x * 256 + threadIdx.x) >> 5;
  int lane = threadIdx.x & 31;
  if (b >= SIZE_B) return;
  float* th = tout + (size_t)b * TEXT_S * DMODEL;     // token 0 row
  const float* ghr = gh + (size_t)b * DMODEL;
  const float* tsr = ts + (size_t)b * DMODEL;
  const float* gsr = gs + (size_t)b * DMODEL;
  float att = 0, atg = 0, agt = 0, agg = 0;
  float thv[8], ghv[8], tsv[8], gsv[8];
#pragma unroll
  for (int i = 0; i < 8; ++i) {
    int c = lane + i * 32;
    thv[i] = th[c]; ghv[i] = ghr[c]; tsv[i] = tsr[c]; gsv[i] = gsr[c];
    att += thv[i] * tsv[i]; atg += thv[i] * gsv[i];
    agt += ghv[i] * tsv[i]; agg += ghv[i] * gsv[i];
  }
#pragma unroll
  for (int o = 16; o > 0; o >>= 1) {      // wave32 butterfly reduction
    att += __shfl_xor(att, o, 32);
    atg += __shfl_xor(atg, o, 32);
    agt += __shfl_xor(agt, o, 32);
    agg += __shfl_xor(agg, o, 32);
  }
  float m1 = fmaxf(att, atg);
  float e0 = __expf(att - m1), e1 = __expf(atg - m1);
  float s10 = e0 / (e0 + e1), s11 = e1 / (e0 + e1);
  float m2 = fmaxf(agt, agg);
  float f0 = __expf(agt - m2), f1 = __expf(agg - m2);
  float s20 = f0 / (f0 + f1), s21 = f1 / (f0 + f1);
#pragma unroll
  for (int i = 0; i < 8; ++i) {
    int c = lane + i * 32;
    float tn = s10 * thv[i] + s11 * ghv[i];
    float gn = s20 * tn + s21 * ghv[i];
    th[c] = tn;
    gout[(size_t)b * DMODEL + c] = gn;
  }
}

// ---------------------------------------------------------------------------
extern "C" void kernel_launch(void* const* d_in, const int* in_sizes, int n_in,
                              void* d_out, int out_size, void* d_ws,
                              size_t ws_size, hipStream_t stream) {
  (void)in_sizes; (void)n_in; (void)out_size; (void)ws_size;
  const float* text   = (const float*)d_in[0];
  const unsigned char* pmask = (const unsigned char*)d_in[1];
  const float* graph  = (const float*)d_in[2];
  const int*   eidx   = (const int*)d_in[3];
  const float* t_in_w = (const float*)d_in[6];
  const float* t_in_b = (const float*)d_in[7];
  const float* t_out_w= (const float*)d_in[8];
  const float* t_out_b= (const float*)d_in[9];
  const float* gcn_w  = (const float*)d_in[10];
  const float* gcn_b  = (const float*)d_in[11];
  const float* g_in_w = (const float*)d_in[12];
  const float* g_in_b = (const float*)d_in[13];
  const float* g_out_w= (const float*)d_in[14];
  const float* g_out_b= (const float*)d_in[15];
  const float* it_w   = (const float*)d_in[16];
  const float* ig_w   = (const float*)d_in[17];

  float* out = (float*)d_out;
  float* text_out  = out;                                        // (B*96,256)
  float* graph_out = out + (size_t)SIZE_B * TEXT_S * DMODEL;     // (32768,256)
  float* attn_out  = graph_out + (size_t)NNODES * DMODEL;        // (B,96,96)

  float* ws = (float*)d_ws;
  size_t off = 0;
  float* qkvA  = ws + off; off += (size_t)SIZE_B * TEXT_S * 3 * DMODEL; // reused for graph qkv
  float* ctx   = ws + off; off += (size_t)SIZE_B * TEXT_S * DMODEL;     // reused for graph ctx
  float* xw    = ws + off; off += (size_t)NNODES * DMODEL;
  float* res   = ws + off; off += (size_t)NNODES * DMODEL;
  float* deg   = ws + off; off += (size_t)NNODES;
  float* neigh = ws + off; off += (size_t)SIZE_B * GS * DMODEL;
  float* gh    = ws + off; off += (size_t)SIZE_B * DMODEL;
  float* ts    = ws + off; off += (size_t)SIZE_B * DMODEL;
  float* gs    = ws + off; off += (size_t)SIZE_B * DMODEL;

  dim3 blk(256);

  // 1) text QKV projection: (196608,256) x (768,256)^T
  {
    int M = SIZE_B * TEXT_S, N = 3 * DMODEL, K = DMODEL;
    gemm_bf16_wmma<<<dim3(N / 64, M / 128), blk, 0, stream>>>(
        text, K, t_in_w, t_in_b, qkvA, N, M, N, K);
  }
  // 2) text attention (masked, writes head-mean attn map)
  mha_kernel<TEXT_S, TEXT_S, true, true>
      <<<SIZE_B, blk, 0, stream>>>(qkvA, pmask, ctx, attn_out);
  // 3) text out projection -> d_out text region (full text_o)
  {
    int M = SIZE_B * TEXT_S, N = DMODEL, K = DMODEL;
    gemm_bf16_wmma<<<dim3(N / 64, M / 128), blk, 0, stream>>>(
        ctx, K, t_out_w, t_out_b, text_out, N, M, N, K);
  }
  // 4) GCN: xw = graph @ gcn_w^T
  {
    int M = NNODES, N = DMODEL, K = DMODEL;
    gemm_bf16_wmma<<<dim3(N / 64, M / 128), blk, 0, stream>>>(
        graph, K, gcn_w, nullptr, xw, N, M, N, K);
  }
  // 5) degrees + normalized aggregation
  deg_init_kernel<<<NNODES / 256, blk, 0, stream>>>(deg);
  deg_count_kernel<<<EFULL / 256, blk, 0, stream>>>(eidx, deg);
  res_base_kernel<<<(NNODES * DMODEL) / 256, blk, 0, stream>>>(xw, deg, gcn_b, res);
  scatter_kernel<<<(EFULL * 32) / 256, blk, 0, stream>>>(eidx, xw, deg, res);
  // 6) neighborhood gather -> (B*33, 256)
  {
    size_t n = (size_t)SIZE_B * GS * DMODEL;
    gather_kernel<<<(unsigned)((n + 255) / 256), blk, 0, stream>>>(res, eidx, neigh);
  }
  // 7) graph QKV projection (reuse qkvA)
  {
    int M = SIZE_B * GS, N = 3 * DMODEL, K = DMODEL;   // 67584 = 128*528
    gemm_bf16_wmma<<<dim3(N / 64, M / 128), blk, 0, stream>>>(
        neigh, K, g_in_w, g_in_b, qkvA, N, M, N, K);
  }
  // 8) graph attention, S=33 padded to 48 (reuse ctx)
  mha_kernel<GS, GST, false, false>
      <<<SIZE_B, blk, 0, stream>>>(qkvA, nullptr, ctx, nullptr);
  // 9) graph out projection, token-0 rows only (lda = 33*256)
  {
    int M = SIZE_B, N = DMODEL, K = DMODEL;
    gemm_bf16_wmma<<<dim3(N / 64, M / 128), blk, 0, stream>>>(
        ctx, GS * DMODEL, g_out_w, g_out_b, gh, N, M, N, K);
  }
  // 10) res[size:] -> graph_out tail
  {
    size_t n = (size_t)(NNODES - SIZE_B) * DMODEL;
    copy_tail_kernel<<<(unsigned)((n + 255) / 256), blk, 0, stream>>>(res, graph_out);
  }
  // 11) interlayer scores (th read strided from d_out token-0 rows)
  gemm_bf16_wmma<<<dim3(DMODEL / 64, SIZE_B / 128), blk, 0, stream>>>(
      text_out, TEXT_S * DMODEL, it_w, nullptr, ts, DMODEL, SIZE_B, DMODEL, DMODEL);
  gemm_bf16_wmma<<<dim3(DMODEL / 64, SIZE_B / 128), blk, 0, stream>>>(
      gh, DMODEL, ig_w, nullptr, gs, DMODEL, SIZE_B, DMODEL, DMODEL);
  // 12) gating -> overwrite token-0 text rows + graph_out head rows
  gate_kernel<<<(SIZE_B * 32) / 256, blk, 0, stream>>>(text_out, gh, ts, gs, graph_out);
}
